// SelfAttention_65790309040498
// MI455X (gfx1250) — matrix operand order repaired and checked by reading, compiled, not run on hardware
//
#include <hip/hip_runtime.h>

// ---------------------------------------------------------------------------
// Self-attention forward for MI455X (gfx1250, wave32, WMMA).
//   hidden:  [2,2048,1024] f32       W*: [1024,1024] f32   b*: [1024] f32
//   out:     ctx [2,2048,1024] f32  ++  probs [2,16,2048,2048] f32
// Math in bf16 WMMA (v_wmma_f32_16x16x32_bf16), fp32 accumulate.
// Scores+softmax+ctx fused per q-strip so probs hits HBM exactly once.
// All WMMA k-loops software-pipelined (prefetch next fragments before MMA).
// ---------------------------------------------------------------------------

#define HIDDEN 1024
#define NHEADS 16
#define HDIM   64
#define BATCH  2
#define SEQ    2048
#define BS     (BATCH * SEQ)      // 4096 tokens
#define SCALE  0.125f             // 1/sqrt(64)

typedef __attribute__((ext_vector_type(16))) __bf16 v16bf;
typedef __attribute__((ext_vector_type(8)))  __bf16 v8bf;
typedef __attribute__((ext_vector_type(8)))  float  v8f;
typedef __attribute__((ext_vector_type(4)))  float  v4f;

static __device__ __forceinline__ __bf16 f2bf16(float f) {
    unsigned u = __builtin_bit_cast(unsigned, f);
    u += 0x7FFFu + ((u >> 16) & 1u);            // round-to-nearest-even
    unsigned short h = (unsigned short)(u >> 16);
    return __builtin_bit_cast(__bf16, h);
}

// ---- WMMA fragment loaders (ISA 7.12.2 16-bit layouts) ---------------------
// A (16x32, MxK): lane L -> row m=L&15; half=(L>>4): K = k0+half*8+{0..7} and
//                 k0+16+half*8+{0..7}
static __device__ __forceinline__ v16bf load_frag_a_bf16(
        const __bf16* __restrict__ base, int ld, int k0, int lane) {
    int m    = lane & 15;
    int half = (lane >> 4) & 1;
    const __bf16* p = base + (size_t)m * ld + k0 + half * 8;
    v8bf lo = *(const v8bf*)(p);
    v8bf hi = *(const v8bf*)(p + 16);
    v16bf r;
#pragma unroll
    for (int i = 0; i < 8; ++i) { r[i] = lo[i]; r[i + 8] = hi[i]; }
    return r;
}

// B (32x16, KxN): lane L -> col n=L&15; half=(L>>4): K = k0+half*8+{0..7} and
//                 k0+16+half*8+{0..7}.
// base points at "row n" = the K-contiguous vector feeding column n.
static __device__ __forceinline__ v16bf load_frag_b_bf16(
        const __bf16* __restrict__ base, int ld, int k0, int lane) {
    int n    = lane & 15;
    int half = (lane >> 4) & 1;
    const __bf16* p = base + (size_t)n * ld + k0 + half * 8;
    v8bf lo = *(const v8bf*)(p);
    v8bf hi = *(const v8bf*)(p + 16);
    v16bf r;
#pragma unroll
    for (int i = 0; i < 8; ++i) { r[i] = lo[i]; r[i + 8] = hi[i]; }
    return r;
}

static __device__ __forceinline__ v8f wmma_bf16(v16bf a, v16bf b, v8f c) {
    return __builtin_amdgcn_wmma_f32_16x16x32_bf16(
        /*neg_a=*/false, a, /*neg_b=*/false, b,
        /*c_mod=*/(short)0, c, /*reuse_a=*/false, /*reuse_b=*/false);
}

// ---------------------------------------------------------------------------
// Kernel 0: fp32 -> bf16 bulk convert (n multiple of 4)
// ---------------------------------------------------------------------------
__global__ void __launch_bounds__(256)
cvt_f32_bf16(const float* __restrict__ src, __bf16* __restrict__ dst, int n) {
    int i = (blockIdx.x * blockDim.x + threadIdx.x) * 4;
    if (i >= n) return;
    v4f x = *(const v4f*)(src + i);
    dst[i + 0] = f2bf16(x.x);
    dst[i + 1] = f2bf16(x.y);
    dst[i + 2] = f2bf16(x.z);
    dst[i + 3] = f2bf16(x.w);
}

// ---------------------------------------------------------------------------
// Kernel 1: fused QKV projection.  One wave = one 16x16 (m,n) tile; one A
// fragment feeds three WMMAs per k-step (Wq/Wk/Wv).  Software-pipelined:
// k+1 fragments are in flight across the k-step WMMAs.  Q,K row-major bf16;
// V stored transposed [b,h,d,s] so the ctx GEMM gets K-contiguous B rows.
// waves = (4096/16)*(1024/16) = 16384 -> 2048 blocks of 8 waves
// ---------------------------------------------------------------------------
__global__ void __launch_bounds__(256)
qkv_gemm(const __bf16* __restrict__ X,
         const __bf16* __restrict__ Wq, const __bf16* __restrict__ Wk,
         const __bf16* __restrict__ Wv,
         const float* __restrict__ bq, const float* __restrict__ bk,
         const float* __restrict__ bv,
         __bf16* __restrict__ Q, __bf16* __restrict__ K,
         __bf16* __restrict__ Vt) {
    int wave = (blockIdx.x * blockDim.x + threadIdx.x) >> 5;
    int lane = threadIdx.x & 31;
    int mt = wave >> 6;           // 0..255 (token tile)
    int nt = wave & 63;           // 0..63  (feature tile)

    const __bf16* Abase  = X  + (size_t)mt * 16 * HIDDEN;
    const __bf16* Bqbase = Wq + (size_t)nt * 16 * HIDDEN;
    const __bf16* Bkbase = Wk + (size_t)nt * 16 * HIDDEN;
    const __bf16* Bvbase = Wv + (size_t)nt * 16 * HIDDEN;

    v8f accq = {}, acck = {}, accv = {};

    // prologue: fragments for k0 = 0
    v16bf a  = load_frag_a_bf16(Abase,  HIDDEN, 0, lane);
    v16bf fq = load_frag_b_bf16(Bqbase, HIDDEN, 0, lane);
    v16bf fk = load_frag_b_bf16(Bkbase, HIDDEN, 0, lane);
    v16bf fv = load_frag_b_bf16(Bvbase, HIDDEN, 0, lane);

#pragma unroll 4
    for (int k0 = 0; k0 < HIDDEN - 32; k0 += 32) {
        // prefetch k0+32 before consuming k0 -> loads overlap WMMAs
        v16bf a_n  = load_frag_a_bf16(Abase,  HIDDEN, k0 + 32, lane);
        v16bf fq_n = load_frag_b_bf16(Bqbase, HIDDEN, k0 + 32, lane);
        v16bf fk_n = load_frag_b_bf16(Bkbase, HIDDEN, k0 + 32, lane);
        v16bf fv_n = load_frag_b_bf16(Bvbase, HIDDEN, k0 + 32, lane);
        accq = wmma_bf16(a, fq, accq);
        acck = wmma_bf16(a, fk, acck);
        accv = wmma_bf16(a, fv, accv);
        a = a_n; fq = fq_n; fk = fk_n; fv = fv_n;
    }
    accq = wmma_bf16(a, fq, accq);
    acck = wmma_bf16(a, fk, acck);
    accv = wmma_bf16(a, fv, accv);

    int col  = lane & 15;
    int half = (lane >> 4) & 1;
    int n = nt * 16 + col;                 // global output feature
    float vbq = bq[n], vbk = bk[n], vbv = bv[n];
    int h = n >> 6, d = n & 63;
#pragma unroll
    for (int i = 0; i < 8; ++i) {
        int m = mt * 16 + i + half * 8;    // global token row 0..4095
        Q[(size_t)m * HIDDEN + n] = f2bf16(accq[i] + vbq);
        K[(size_t)m * HIDDEN + n] = f2bf16(acck[i] + vbk);
        int b = m >> 11, s = m & (SEQ - 1);
        Vt[((size_t)(b * NHEADS + h) * HDIM + d) * SEQ + s] =
            f2bf16(accv[i] + vbv);
    }
}

// ---------------------------------------------------------------------------
// Kernel 2: FUSED scores + softmax + ctx, one workgroup per (b,h,q-strip).
//
//   8 waves; wave w owns score cols [w*256, w*256+256) of a 16x2048 strip:
//     16 tiles x 2 WMMAs, accumulators live in 128 VGPRs, K fragments for
//     tile t+1 prefetched across tile t's WMMAs.
//   Softmax max/sum via 8KB LDS reductions (no HBM round trip).
//   probs written to HBM exactly once (fp32); simultaneously staged bf16 in
//   a 64KB LDS strip P[16][2048] -- doubles as D-layout -> A-layout
//   transpose.
//   ctx: waves split as (n-tile = w&3, k-half = w>>2); A fragments via
//   ds_load_b128 from P, B from transposed V, both prefetched one step
//   ahead; k-half partials merged in LDS.
//
// blocks = 2*16*128 = 4096, 256 threads
// ---------------------------------------------------------------------------
__global__ void __launch_bounds__(256)
fused_attn(const __bf16* __restrict__ Q, const __bf16* __restrict__ K,
           const __bf16* __restrict__ Vt,
           float* __restrict__ probs, float* __restrict__ ctx) {
    __shared__ __bf16 P[16][SEQ];          // 64 KiB probs strip (bf16)
    __shared__ float  lred[8][32][8];      // 8 KiB reduction / partial buffer
    __shared__ float  rowstat[16];

    int idx = blockIdx.x;
    int qt  = idx & 127;
    int bh  = idx >> 7;                    // 0..31
    int b   = bh >> 4, h = bh & 15;
    int w    = threadIdx.x >> 5;
    int lane = threadIdx.x & 31;
    int col  = lane & 15;
    int half = (lane >> 4) & 1;

    // ---- Phase 1: scores strip (pipelined over 16 key tiles) -------------
    const __bf16* Qbase = Q + ((size_t)(b * SEQ + qt * 16)) * HIDDEN + h * HDIM;
    const __bf16* Kbase = K + ((size_t)(b * SEQ + w * 256)) * HIDDEN + h * HDIM;

    v16bf aq0 = load_frag_a_bf16(Qbase, HIDDEN, 0,  lane);
    v16bf aq1 = load_frag_a_bf16(Qbase, HIDDEN, 32, lane);

    v16bf f0 = load_frag_b_bf16(Kbase, HIDDEN, 0,  lane);
    v16bf f1 = load_frag_b_bf16(Kbase, HIDDEN, 32, lane);

    v8f sc[16];
#pragma unroll
    for (int t = 0; t < 16; ++t) {
        v16bf f0n = {}, f1n = {};
        if (t < 15) {
            const __bf16* Bn = Kbase + (size_t)(t + 1) * 16 * HIDDEN;
            f0n = load_frag_b_bf16(Bn, HIDDEN, 0,  lane);
            f1n = load_frag_b_bf16(Bn, HIDDEN, 32, lane);
        }
        v8f acc = {};
        acc = wmma_bf16(aq0, f0, acc);
        acc = wmma_bf16(aq1, f1, acc);
        sc[t] = acc;
        f0 = f0n; f1 = f1n;
    }

    // ---- Phase 2: row max (on raw scores; scale folded into exp) ---------
    float pmax[8];
#pragma unroll
    for (int i = 0; i < 8; ++i) pmax[i] = sc[0][i];
#pragma unroll
    for (int t = 1; t < 16; ++t)
#pragma unroll
        for (int i = 0; i < 8; ++i) pmax[i] = fmaxf(pmax[i], sc[t][i]);
#pragma unroll
    for (int i = 0; i < 8; ++i) lred[w][lane][i] = pmax[i];
    __syncthreads();
    if (threadIdx.x < 16) {
        int r = threadIdx.x, hh = r >> 3, ii = r & 7;
        float m = -3.402823466e38f;
        for (int w2 = 0; w2 < 8; ++w2)
            for (int l = hh * 16; l < hh * 16 + 16; ++l)
                m = fmaxf(m, lred[w2][l][ii]);
        rowstat[r] = m;
    }
    __syncthreads();

    // ---- Phase 3: exp + row sum ------------------------------------------
    float rm[8], psum[8];
#pragma unroll
    for (int i = 0; i < 8; ++i) { rm[i] = rowstat[i + half * 8]; psum[i] = 0.f; }
#pragma unroll
    for (int t = 0; t < 16; ++t)
#pragma unroll
        for (int i = 0; i < 8; ++i) {
            float e = __expf((sc[t][i] - rm[i]) * SCALE);
            sc[t][i] = e;
            psum[i] += e;
        }
#pragma unroll
    for (int i = 0; i < 8; ++i) lred[w][lane][i] = psum[i];
    __syncthreads();
    if (threadIdx.x < 16) {
        int r = threadIdx.x, hh = r >> 3, ii = r & 7;
        float s = 0.f;
        for (int w2 = 0; w2 < 8; ++w2)
            for (int l = hh * 16; l < hh * 16 + 16; ++l)
                s += lred[w2][l][ii];
        rowstat[r] = 1.0f / s;
    }
    __syncthreads();
    float inv[8];
#pragma unroll
    for (int i = 0; i < 8; ++i) inv[i] = rowstat[i + half * 8];

    // ---- Phase 4: write probs (HBM, once) + stage bf16 strip in LDS ------
    float* prow = probs + ((size_t)bh * SEQ + (size_t)qt * 16) * SEQ;
#pragma unroll
    for (int t = 0; t < 16; ++t) {
        int c = (w * 16 + t) * 16 + col;
#pragma unroll
        for (int i = 0; i < 8; ++i) {
            int r  = i + half * 8;
            float v = sc[t][i] * inv[i];
            prow[(size_t)r * SEQ + c] = v;
            P[r][c] = f2bf16(v);
        }
    }
    __syncthreads();

    // ---- Phase 5: ctx = P @ V  (n-tile = w&3, k-half = w>>2), pipelined --
    int ntile = w & 3;
    int khalf = w >> 2;
    const __bf16* Bb = Vt + ((size_t)bh * HDIM + ntile * 16) * SEQ;

    int kbeg = khalf * (SEQ / 2);
    int kend = kbeg + (SEQ / 2);

    // prologue
    v16bf a;
    {
        const __bf16* pa = &P[col][kbeg + half * 8];
        v8bf lo = *(const v8bf*)(pa);
        v8bf hi = *(const v8bf*)(pa + 16);
#pragma unroll
        for (int i = 0; i < 8; ++i) { a[i] = lo[i]; a[i + 8] = hi[i]; }
    }
    v16bf f = load_frag_b_bf16(Bb, SEQ, kbeg, lane);

    v8f acc = {};
#pragma unroll 4
    for (int k0 = kbeg; k0 < kend - 32; k0 += 32) {
        v16bf a_n;
        {
            const __bf16* pa = &P[col][k0 + 32 + half * 8];
            v8bf lo = *(const v8bf*)(pa);
            v8bf hi = *(const v8bf*)(pa + 16);
#pragma unroll
            for (int i = 0; i < 8; ++i) { a_n[i] = lo[i]; a_n[i + 8] = hi[i]; }
        }
        v16bf f_n = load_frag_b_bf16(Bb, SEQ, k0 + 32, lane);
        acc = wmma_bf16(a, f, acc);
        a = a_n; f = f_n;
    }
    acc = wmma_bf16(a, f, acc);

    float* pbuf = &lred[0][0][0];          // reuse 8KB buffer for partials
    if (w >= 4) {
#pragma unroll
        for (int i = 0; i < 8; ++i)
            pbuf[(((w - 4) * 32) + lane) * 8 + i] = acc[i];
    }
    __syncthreads();
    if (w < 4) {
        int dcol = h * HDIM + ntile * 16 + col;
#pragma unroll
        for (int i = 0; i < 8; ++i) {
            float v = acc[i] + pbuf[((w * 32) + lane) * 8 + i];
            int s = qt * 16 + i + half * 8;
            ctx[((size_t)(b * SEQ + s)) * HIDDEN + dcol] = v;
        }
    }
}

// ---------------------------------------------------------------------------
// Host side
// ---------------------------------------------------------------------------
extern "C" void kernel_launch(void* const* d_in, const int* in_sizes, int n_in,
                              void* d_out, int out_size, void* d_ws, size_t ws_size,
                              hipStream_t stream) {
    (void)in_sizes; (void)n_in; (void)out_size; (void)ws_size;

    const float* hs = (const float*)d_in[0];
    const float* Wq = (const float*)d_in[1];
    const float* bq = (const float*)d_in[2];
    const float* Wk = (const float*)d_in[3];
    const float* bk = (const float*)d_in[4];
    const float* Wv = (const float*)d_in[5];
    const float* bv = (const float*)d_in[6];

    float* ctx   = (float*)d_out;                       // [2,2048,1024]
    float* probs = ctx + (size_t)BATCH * SEQ * HIDDEN;  // [2,16,2048,2048]

    // Workspace layout (38 MiB total)
    char* ws = (char*)d_ws;
    __bf16* Xb  = (__bf16*)(ws);                        //  8 MiB  X bf16
    __bf16* Wqb = (__bf16*)(ws + ( 8ull << 20));        //  2 MiB
    __bf16* Wkb = (__bf16*)(ws + (10ull << 20));        //  2 MiB
    __bf16* Wvb = (__bf16*)(ws + (12ull << 20));        //  2 MiB
    __bf16* Qb  = (__bf16*)(ws + (14ull << 20));        //  8 MiB
    __bf16* Kb  = (__bf16*)(ws + (22ull << 20));        //  8 MiB
    __bf16* Vtb = (__bf16*)(ws + (30ull << 20));        //  8 MiB  V^T [b,h,d,s]

    // 0) fp32 -> bf16
    cvt_f32_bf16<<<(BS * HIDDEN / 4 + 255) / 256, 256, 0, stream>>>(hs, Xb, BS * HIDDEN);
    cvt_f32_bf16<<<(HIDDEN * HIDDEN / 4 + 255) / 256, 256, 0, stream>>>(Wq, Wqb, HIDDEN * HIDDEN);
    cvt_f32_bf16<<<(HIDDEN * HIDDEN / 4 + 255) / 256, 256, 0, stream>>>(Wk, Wkb, HIDDEN * HIDDEN);
    cvt_f32_bf16<<<(HIDDEN * HIDDEN / 4 + 255) / 256, 256, 0, stream>>>(Wv, Wvb, HIDDEN * HIDDEN);

    // 1) fused QKV projection (bias fused, V transposed per head)
    qkv_gemm<<<2048, 256, 0, stream>>>(Xb, Wqb, Wkb, Wvb, bq, bk, bv, Qb, Kb, Vtb);

    // 2) fused scores + softmax + ctx (probs touches HBM exactly once)
    fused_attn<<<BATCH * NHEADS * (SEQ / 16), 256, 0, stream>>>(Qb, Kb, Vtb,
                                                                probs, ctx);
}
